// ExclusiveSelfAttention_84335977824599
// MI455X (gfx1250) — matrix operand order repaired and checked by reading, not compiled, not run on hardware
//
#include <hip/hip_runtime.h>
#include <hip/hip_bf16.h>

// MI455X / gfx1250 implementation.
// Pipeline: cvt(fp32->bf16) -> 3x WMMA GEMM (Q/K/V proj) -> flash-attn w/
// fused Gram-Schmidt exclusion -> WMMA GEMM (output proj, fp32 out).
// All matmuls use v_wmma_f32_16x16x32_bf16 (fp32 accumulate).
// K tiles staged into LDS via global_load_async_to_lds_b128 (ASYNCcnt).
// Cross-lane softmax reductions use DPP16 ROW_XMASK (VALU) instead of
// ds_bpermute round trips.
// Workspace layout (needs 36 MB):
//   [0,8M)    xb   : x as bf16              [4096,1024]
//   [8M,10M)  wqb  : wq bf16 [out,in]
//   [10M,12M) wkb
//   [12M,14M) wvb
//   [14M,16M) wob
//   [16M,20M) qb   : [B,H,S,64] bf16
//   [20M,24M) kb
//   [24M,28M) vb
//   [28M,36M) ob   : merged-head attn out, bf16 [4096,1024]

typedef __attribute__((ext_vector_type(16))) __bf16 v16bf;
typedef __attribute__((ext_vector_type(8)))  float  v8f;
typedef __attribute__((ext_vector_type(4)))  int    i4v;

union BF16Frag { v16bf v; uint4 u[2]; unsigned short s[16]; };

#define HAS_ASYNC_LDS __has_builtin(__builtin_amdgcn_global_load_async_to_lds_b128)
#define HAS_DPP       __has_builtin(__builtin_amdgcn_update_dpp)

static __device__ __forceinline__ unsigned short f32_to_bf16(float f) {
  unsigned u = __float_as_uint(f);
  u += 0x7FFFu + ((u >> 16) & 1u);          // round-to-nearest-even
  return (unsigned short)(u >> 16);
}
static __device__ __forceinline__ float bf16_to_f32(unsigned short h) {
  return __uint_as_float(((unsigned)h) << 16);
}
static __device__ __forceinline__ v8f wmma_bf16(v16bf a, v16bf b, v8f c) {
  // (neg_a, A, neg_b, B, c_mod, C, reuse_a, reuse_b)
  return __builtin_amdgcn_wmma_f32_16x16x32_bf16(false, a, false, b, (short)0, c,
                                                 false, false);
}

// lane <- lane^MASK within each 16-lane row, as a pipelined VALU op
template <int MASK>
static __device__ __forceinline__ float dpp_xor(float x) {
#if HAS_DPP
  return __int_as_float(__builtin_amdgcn_update_dpp(
      0, __float_as_int(x), 0x160 | MASK /*ROW_XMASK*/, 0xf, 0xf, true));
#else
  return __shfl_xor(x, MASK, 32);
#endif
}

// max-reduce 8 independent channels across the 16 lanes of each half-wave
static __device__ __forceinline__ void red16_max8(float m[8]) {
#pragma unroll
  for (int j = 0; j < 8; ++j) m[j] = fmaxf(m[j], dpp_xor<1>(m[j]));
#pragma unroll
  for (int j = 0; j < 8; ++j) m[j] = fmaxf(m[j], dpp_xor<2>(m[j]));
#pragma unroll
  for (int j = 0; j < 8; ++j) m[j] = fmaxf(m[j], dpp_xor<4>(m[j]));
#pragma unroll
  for (int j = 0; j < 8; ++j) m[j] = fmaxf(m[j], dpp_xor<8>(m[j]));
}
// sum-reduce 8 independent channels across the 16 lanes of each half-wave
static __device__ __forceinline__ void red16_sum8(float m[8]) {
#pragma unroll
  for (int j = 0; j < 8; ++j) m[j] += dpp_xor<1>(m[j]);
#pragma unroll
  for (int j = 0; j < 8; ++j) m[j] += dpp_xor<2>(m[j]);
#pragma unroll
  for (int j = 0; j < 8; ++j) m[j] += dpp_xor<4>(m[j]);
#pragma unroll
  for (int j = 0; j < 8; ++j) m[j] += dpp_xor<8>(m[j]);
}

// ---------------------------------------------------------------- convert
__global__ void cvt_f32_bf16(const float* __restrict__ in,
                             unsigned short* __restrict__ out, int n) {
  int i = blockIdx.x * blockDim.x + threadIdx.x;
  int stride = gridDim.x * blockDim.x;
  for (; i < n; i += stride) out[i] = f32_to_bf16(in[i]);
}

// ---------------------------------------------------------------- GEMM
// C[M=4096, N=1024] = A[M,1024](bf16) * W[N,1024](bf16, torch [out,in]) ^T + bias
// Each wave owns a 32x64 tile: 2 A-fragments share 4 B-fragments per K-slice
// (12 B128 loads : 8 WMMAs).
// MODE 0: store bf16 in [B,H,S,64] head layout (QKV proj)
// MODE 1: store fp32 row-major [M,N] (output proj)
template <int MODE>
__global__ __launch_bounds__(128) void gemm_wmma(
    const unsigned short* __restrict__ A, const unsigned short* __restrict__ W,
    const float* __restrict__ bias, void* __restrict__ outp) {
  const int tid  = threadIdx.x;
  const int lane = tid & 31;
  const int wid  = tid >> 5;
  const int l15  = lane & 15;
  const int half = lane >> 4;
  const int gw    = blockIdx.x * 4 + wid;   // 2048 waves
  const int mbase = (gw >> 4) * 32;         // 128 row tiles of 32
  const int nbase = (gw & 15) * 64;         // 16 col tiles of 64
  const int akb = half ? 16 : 0;            // A frag k-offset per lane half
  const int bkh = half ? 16 : 0;            // B frag k-offset per lane half

  v8f c[2][4] = {};
  const unsigned short* arow0 = A + (mbase + l15) * 1024;
  const unsigned short* arow1 = A + (mbase + 16 + l15) * 1024;

  for (int kk = 0; kk < 1024; kk += 32) {
    __builtin_prefetch(arow0 + kk + 512, 0, 1);  // global_prefetch_b8
    __builtin_prefetch(arow1 + kk + 512, 0, 1);
    BF16Frag a0, a1;
    a0.u[0] = *(const uint4*)(arow0 + kk + akb);
    a0.u[1] = *(const uint4*)(arow0 + kk + akb + 8);
    a1.u[0] = *(const uint4*)(arow1 + kk + akb);
    a1.u[1] = *(const uint4*)(arow1 + kk + akb + 8);
#pragma unroll
    for (int nt = 0; nt < 4; ++nt) {
      const unsigned short* wrow = W + (nbase + nt * 16 + l15) * 1024 + kk + bkh;
      BF16Frag bfg;
      bfg.u[0] = *(const uint4*)(wrow);
      bfg.u[1] = *(const uint4*)(wrow + 8);
      c[0][nt] = wmma_bf16(a0.v, bfg.v, c[0][nt]);
      c[1][nt] = wmma_bf16(a1.v, bfg.v, c[1][nt]);
    }
  }

#pragma unroll
  for (int mi = 0; mi < 2; ++mi) {
#pragma unroll
    for (int nt = 0; nt < 4; ++nt) {
      const int n = nbase + nt * 16 + l15;
      const float bv = bias[n];
#pragma unroll
      for (int j = 0; j < 8; ++j) {
        const int m = mbase + mi * 16 + j + half * 8;
        const float val = c[mi][nt][j] + bv;
        if (MODE == 0) {
          unsigned short* out = (unsigned short*)outp;
          const int b = m >> 11, s = m & 2047;   // S = 2048
          const int h = n >> 6, hd = n & 63;     // HD = 64
          out[(((b * 16 + h) * 2048) + s) * 64 + hd] = f32_to_bf16(val);
        } else {
          float* out = (float*)outp;
          out[m * 1024 + n] = val;
        }
      }
    }
  }
}

// ---------------------------------------------------------------- attention
// One workgroup (4 waves) = 64 query rows of one (b,h). Flash-attention over
// key tiles of 32 with online softmax; epilogue fuses the v-exclusion:
//   o -= (o.v)/(v.v+eps) * v ,   then writes merged-head bf16 [4096,1024].
__global__ __launch_bounds__(128) void attn_wmma(
    const unsigned short* __restrict__ Q, const unsigned short* __restrict__ K,
    const unsigned short* __restrict__ V, unsigned short* __restrict__ O) {
  __shared__ __align__(16) unsigned short sK[32 * 64];   // [key][hd]
  __shared__ __align__(16) unsigned short sVt[64 * 32];  // [hd][key]
  __shared__ __align__(16) unsigned short sP[4 * 16 * 32];

  const int tid  = threadIdx.x;
  const int lane = tid & 31;
  const int wid  = tid >> 5;
  const int l15  = lane & 15;
  const int half = lane >> 4;
  const int bh = blockIdx.x >> 5;   // b*16+h  (32 total)
  const int qt = blockIdx.x & 31;   // 32 query tiles of 64
  const int qbase = qt * 64 + wid * 16;
  const int akb = half ? 16 : 0;
  const int bkh = half ? 16 : 0;

  // preload this wave's Q fragments (16 rows x 64 hd = 2 A-fragments)
  BF16Frag qf[2];
  {
    const unsigned short* qrow = Q + (bh * 2048 + qbase + l15) * 64;
#pragma unroll
    for (int k2 = 0; k2 < 2; ++k2) {
      qf[k2].u[0] = *(const uint4*)(qrow + k2 * 32 + akb);
      qf[k2].u[1] = *(const uint4*)(qrow + k2 * 32 + akb + 8);
    }
  }

  v8f oacc[4] = {};
  v8f rmax, rsum;
#pragma unroll
  for (int j = 0; j < 8; ++j) { rmax[j] = -1e30f; rsum[j] = 0.0f; }

  const unsigned short* ksrc0 = K + bh * 2048 * 64;
  const unsigned short* vsrc0 = V + bh * 2048 * 64;

  for (int t = 0; t < 64; ++t) {
    // ---- stage K tile: async copy straight into LDS when available
    {
#if HAS_ASYNC_LDS
      char* gsrc = (char*)(ksrc0 + t * 32 * 64);
      char* ldst = (char*)sK;
      __builtin_amdgcn_global_load_async_to_lds_b128(
          (i4v*)(gsrc + tid * 16), (i4v*)(ldst + tid * 16), 0, 0);
      __builtin_amdgcn_global_load_async_to_lds_b128(
          (i4v*)(gsrc + 2048 + tid * 16), (i4v*)(ldst + 2048 + tid * 16), 0, 0);
#else
      const uint4* src = (const uint4*)(ksrc0 + t * 32 * 64);
      uint4* dst = (uint4*)sK;
      dst[tid] = src[tid];
      dst[tid + 128] = src[tid + 128];
#endif
      // ---- stage V transposed: 2x B128 reads, 16 b16 LDS stores per thread
      const int key = tid >> 2, hd0 = (tid & 3) * 16;
      const unsigned short* vrow = vsrc0 + t * 32 * 64 + key * 64 + hd0;
      unsigned short tmp[16];
      *(uint4*)(tmp)     = *(const uint4*)(vrow);
      *(uint4*)(tmp + 8) = *(const uint4*)(vrow + 8);
#pragma unroll
      for (int i = 0; i < 16; ++i) sVt[(hd0 + i) * 32 + key] = tmp[i];
#if HAS_ASYNC_LDS
#if __has_builtin(__builtin_amdgcn_s_wait_asynccnt)
      __builtin_amdgcn_s_wait_asynccnt(0);
#else
      asm volatile("s_wait_asynccnt 0" ::: "memory");
#endif
#endif
    }
    __syncthreads();

    // scores: Q(16x64) * K^T(64x32) -> two 16x16 fp32 fragments
    v8f s0 = {}, s1 = {};
#pragma unroll
    for (int k2 = 0; k2 < 2; ++k2) {
      const unsigned short* p0 = sK + l15 * 64 + k2 * 32 + bkh;
      const unsigned short* p1 = sK + (16 + l15) * 64 + k2 * 32 + bkh;
      BF16Frag b0, b1;
      b0.u[0] = *(const uint4*)(p0); b0.u[1] = *(const uint4*)(p0 + 8);
      b1.u[0] = *(const uint4*)(p1); b1.u[1] = *(const uint4*)(p1 + 8);
      s0 = wmma_bf16(qf[k2].v, b0.v, s0);
      s1 = wmma_bf16(qf[k2].v, b1.v, s1);
    }

    // online softmax (scale 1/sqrt(64) = 0.125); DPP16 ROW_XMASK reductions
    float mx[8];
#pragma unroll
    for (int j = 0; j < 8; ++j) {
      s0[j] *= 0.125f; s1[j] *= 0.125f;
      mx[j] = fmaxf(s0[j], s1[j]);
    }
    red16_max8(mx);
    float ps[8];
#pragma unroll
    for (int j = 0; j < 8; ++j) {
      const float nmax = fmaxf(rmax[j], mx[j]);
      const float corr = __expf(rmax[j] - nmax);
      rmax[j] = nmax;
      s0[j] = __expf(s0[j] - nmax);
      s1[j] = __expf(s1[j] - nmax);
      ps[j] = s0[j] + s1[j];
      rsum[j] *= corr;
#pragma unroll
      for (int nt = 0; nt < 4; ++nt) oacc[nt][j] *= corr;
    }
    red16_sum8(ps);
#pragma unroll
    for (int j = 0; j < 8; ++j) rsum[j] += ps[j];

    // C-layout -> A-layout for P via per-wave LDS staging
    unsigned short* pw = sP + wid * 512;
#pragma unroll
    for (int j = 0; j < 8; ++j) {
      const int m = j + half * 8;
      pw[m * 32 + l15]      = f32_to_bf16(s0[j]);
      pw[m * 32 + 16 + l15] = f32_to_bf16(s1[j]);
    }
    asm volatile("s_wait_dscnt 0" ::: "memory");  // wave-local LDS RAW

    BF16Frag pf;
    pf.u[0] = *(const uint4*)(pw + l15 * 32 + akb);
    pf.u[1] = *(const uint4*)(pw + l15 * 32 + akb + 8);

    // O += P(16x32) * V(32x64)
#pragma unroll
    for (int nt = 0; nt < 4; ++nt) {
      const unsigned short* vp = sVt + (nt * 16 + l15) * 32 + bkh;
      BF16Frag vb;
      vb.u[0] = *(const uint4*)(vp); vb.u[1] = *(const uint4*)(vp + 8);
      oacc[nt] = wmma_bf16(pf.v, vb.v, oacc[nt]);
    }
    __syncthreads();
  }

  // normalize, then fused Gram-Schmidt exclusion + merged-head store
#pragma unroll
  for (int nt = 0; nt < 4; ++nt)
#pragma unroll
    for (int j = 0; j < 8; ++j) oacc[nt][j] *= 1.0f / rsum[j];

  const int b = bh >> 4, h = bh & 15;
  float vf[4][8], ov[8], vv[8];
#pragma unroll
  for (int j = 0; j < 8; ++j) {
    const int srow = qbase + j + half * 8;
    const unsigned short* vrow = V + (bh * 2048 + srow) * 64;
    ov[j] = 0.0f; vv[j] = 0.0f;
#pragma unroll
    for (int nt = 0; nt < 4; ++nt) {
      const float xv = bf16_to_f32(vrow[nt * 16 + l15]);
      vf[nt][j] = xv;
      ov[j] += oacc[nt][j] * xv;
      vv[j] += xv * xv;
    }
  }
  red16_sum8(ov);
  red16_sum8(vv);
#pragma unroll
  for (int j = 0; j < 8; ++j) {
    const int srow = qbase + j + half * 8;
    const float alignc = ov[j] / (vv[j] + 1e-8f);
#pragma unroll
    for (int nt = 0; nt < 4; ++nt) {
      const float o = oacc[nt][j] - alignc * vf[nt][j];
      O[(b * 2048 + srow) * 1024 + h * 64 + nt * 16 + l15] = f32_to_bf16(o);
    }
  }
}

// ---------------------------------------------------------------- launch
extern "C" void kernel_launch(void* const* d_in, const int* in_sizes, int n_in,
                              void* d_out, int out_size, void* d_ws, size_t ws_size,
                              hipStream_t stream) {
  const float* x  = (const float*)d_in[0];
  const float* wq = (const float*)d_in[1];
  const float* bq = (const float*)d_in[2];
  const float* wk = (const float*)d_in[3];
  const float* bk = (const float*)d_in[4];
  const float* wv = (const float*)d_in[5];
  const float* bv = (const float*)d_in[6];
  const float* wo = (const float*)d_in[7];
  const float* bo = (const float*)d_in[8];

  char* ws = (char*)d_ws;
  unsigned short* xb  = (unsigned short*)(ws);
  unsigned short* wqb = (unsigned short*)(ws + (size_t)( 8u << 20));
  unsigned short* wkb = (unsigned short*)(ws + (size_t)(10u << 20));
  unsigned short* wvb = (unsigned short*)(ws + (size_t)(12u << 20));
  unsigned short* wob = (unsigned short*)(ws + (size_t)(14u << 20));
  unsigned short* qb  = (unsigned short*)(ws + (size_t)(16u << 20));
  unsigned short* kb  = (unsigned short*)(ws + (size_t)(20u << 20));
  unsigned short* vb  = (unsigned short*)(ws + (size_t)(24u << 20));
  unsigned short* ob  = (unsigned short*)(ws + (size_t)(28u << 20));

  cvt_f32_bf16<<<1024, 256, 0, stream>>>(x,  xb,  4096 * 1024);
  cvt_f32_bf16<<<512,  256, 0, stream>>>(wq, wqb, 1024 * 1024);
  cvt_f32_bf16<<<512,  256, 0, stream>>>(wk, wkb, 1024 * 1024);
  cvt_f32_bf16<<<512,  256, 0, stream>>>(wv, wvb, 1024 * 1024);
  cvt_f32_bf16<<<512,  256, 0, stream>>>(wo, wob, 1024 * 1024);

  gemm_wmma<0><<<512, 128, 0, stream>>>(xb, wqb, bq, qb);
  gemm_wmma<0><<<512, 128, 0, stream>>>(xb, wkb, bk, kb);
  gemm_wmma<0><<<512, 128, 0, stream>>>(xb, wvb, bv, vb);

  attn_wmma<<<1024, 128, 0, stream>>>(qb, kb, vb, ob);

  gemm_wmma<1><<<512, 128, 0, stream>>>(ob, wob, bo, d_out);
}
